// SelfAttention_75969381532387
// MI455X (gfx1250) — matrix operand order repaired and checked by reading, not compiled, not run on hardware
//
#include <hip/hip_runtime.h>

typedef _Float16 v16h __attribute__((ext_vector_type(16)));
typedef _Float16 v8h  __attribute__((ext_vector_type(8)));
typedef _Float16 v4h  __attribute__((ext_vector_type(4)));
typedef float    v8f  __attribute__((ext_vector_type(8)));
typedef float    v4f  __attribute__((ext_vector_type(4)));

#define DMODEL 1024
#define HEADS  16
#define DK     64
#define NB     4
#define SEQ    2048
#define MROWS  (NB * SEQ)   // 8192

// ---------------------------------------------------------------------------
// WMMA helper: D = A(16x32 f16) x B(32x16 f16) + C(16x16 f32)
// ---------------------------------------------------------------------------
static __device__ __forceinline__ v8f wmma_f16(v16h a, v16h b, v8f c) {
    return __builtin_amdgcn_wmma_f32_16x16x32_f16(
        /*neg_a=*/false, a, /*neg_b=*/false, b,
        /*c_mod=*/(short)0, c, /*reuse_a=*/false, /*reuse_b=*/false);
}

static __device__ __forceinline__ v16h combine8(v8h lo, v8h hi) {
    return __builtin_shufflevector(lo, hi, 0,1,2,3,4,5,6,7,8,9,10,11,12,13,14,15);
}

// A-operand per-lane layout (ISA 7.12.2, 16-bit A 16x32):
// lanes 0-15 hold K={0..7,16..23}, lanes 16-31 hold K={8..15,24..31}.
// elems[0..7]  = p[half*8 + j], elems[8..15] = p[16 + half*8 + j]
static __device__ __forceinline__ v16h load_a_f16(const _Float16* p, int half) {
    v8h lo = *(const v8h*)(p + half * 8);
    v8h hi = *(const v8h*)(p + 16 + half * 8);
    return combine8(lo, hi);
}

// B-operand: lane = column n (row of the k-major source), same K order as A:
// elems[0..7] = p[half*8 + j], elems[8..15] = p[16 + half*8 + j].
static __device__ __forceinline__ v16h load_b_f16(const _Float16* p, int half) {
    v8h lo = *(const v8h*)(p + half * 8);
    v8h hi = *(const v8h*)(p + 16 + half * 8);
    return combine8(lo, hi);
}

// ---------------------------------------------------------------------------
// Kernel 1: fp32 -> f16 conversion, exact float4 cover (weights + activations)
// ---------------------------------------------------------------------------
__global__ void cvt_f16_kernel(const float* __restrict__ in, _Float16* __restrict__ out) {
    int i = blockIdx.x * blockDim.x + threadIdx.x;   // over float4s
    v4f f = ((const v4f*)in)[i];
    v4h h;
#pragma unroll
    for (int j = 0; j < 4; ++j) h[j] = (_Float16)f[j];
    ((v4h*)out)[i] = h;
}

// ---------------------------------------------------------------------------
// Kernel 2: unified GEMM  Out = A(f16) @ W^T(f16) + b
//   mode 0: f16 out,  head layout  [(n*H+h)*SEQ + l]*DK + d   (Q, K)
//   mode 1: f16 out,  V-transposed [(n*H+h)*DK + d]*SEQ + l   (V)
//   mode 2: f32 out,  row-major    [m*DMODEL + col]           (final proj)
// 8 waves/WG, each wave a 32x64 tile (2 A x 4 B, 8 acc); WG tile 128x128.
// ---------------------------------------------------------------------------
__global__ void linear_kernel(const _Float16* __restrict__ A,
                              const _Float16* __restrict__ Wh,
                              const float* __restrict__ bias,
                              _Float16* __restrict__ out_h,
                              float* __restrict__ out_f,
                              int mode) {
    const int tid  = threadIdx.x;
    const int w    = tid >> 5, lane = tid & 31;
    const int half = lane >> 4, ln = lane & 15;
    const int rowbase = blockIdx.x * 128 + (w >> 1) * 32;
    const int colbase = blockIdx.y * 128 + (w & 1) * 64;

    v8f c[2][4] = {};
    for (int k = 0; k < DMODEL; k += 32) {
        v16h a0 = load_a_f16(A + (size_t)(rowbase + ln)      * DMODEL + k, half);
        v16h a1 = load_a_f16(A + (size_t)(rowbase + 16 + ln) * DMODEL + k, half);
#pragma unroll
        for (int jt = 0; jt < 4; ++jt) {
            v16h b = load_b_f16(Wh + (size_t)(colbase + jt * 16 + ln) * DMODEL + k, half);
            c[0][jt] = wmma_f16(a0, b, c[0][jt]);
            c[1][jt] = wmma_f16(a1, b, c[1][jt]);
        }
    }

#pragma unroll
    for (int jt = 0; jt < 4; ++jt) {
        int col = colbase + jt * 16 + ln;
        float bb = bias[col];
        int h = col >> 6, d = col & (DK - 1);
#pragma unroll
        for (int it = 0; it < 2; ++it) {
#pragma unroll
            for (int v = 0; v < 8; ++v) {
                int m   = rowbase + it * 16 + v + 8 * half;   // C/D: M = v + 8*half
                float val = c[it][jt][v] + bb;
                if (mode == 2) {
                    out_f[(size_t)m * DMODEL + col] = val;
                } else {
                    int nbi = m >> 11, l = m & (SEQ - 1);
                    size_t idx = (mode == 1)
                        ? ((size_t)((nbi * HEADS + h) * DK + d) * SEQ + l)
                        : ((size_t)((nbi * HEADS + h) * SEQ + l) * DK + d);
                    out_h[idx] = (_Float16)val;
                }
            }
        }
    }
}

// ---------------------------------------------------------------------------
// Kernel 3: flash attention. One block per (n, h, 128-q-rows); each of the
// 8 waves owns a 16-row Q tile and streams keys in blocks of 32.
// ---------------------------------------------------------------------------
__global__ void attention_kernel(const _Float16* __restrict__ Qh,
                                 const _Float16* __restrict__ Kh,
                                 const _Float16* __restrict__ Vt,
                                 const int* __restrict__ mask,
                                 _Float16* __restrict__ AO) {
    __shared__ __align__(16) _Float16 pbuf[8][16][40];   // P bounce, padded rows

    const int tid  = threadIdx.x;
    const int w    = tid >> 5, lane = tid & 31;
    const int half = lane >> 4, ln = lane & 15;

    const int b     = blockIdx.x;
    const int qtile = b & 15;          // L/128 = 16 tiles
    const int nh    = b >> 4;          // 0..63
    const int h     = nh & (HEADS - 1);
    const int nb    = nh >> 4;
    const size_t qk_base = (size_t)nh * SEQ * DK;   // [n,h,l,dk]
    const size_t vt_base = (size_t)nh * DK * SEQ;   // [n,h,dk,l]
    const int q0 = qtile * 128 + w * 16;

    // Q tile as WMMA-A, two dk-chunks of 32
    const _Float16* qp = Qh + qk_base + (size_t)(q0 + ln) * DK;
    v16h aq0 = load_a_f16(qp, half);
    v16h aq1 = load_a_f16(qp + 32, half);

    v8f acc[4] = {};
    float mrow[8], lrow[8];
#pragma unroll
    for (int v = 0; v < 8; ++v) { mrow[v] = -3.0e38f; lrow[v] = 0.f; }

    const int* mbase = mask + nb * SEQ;

    for (int kb = 0; kb < SEQ; kb += 32) {
        const _Float16* kp0 = Kh + qk_base + (size_t)(kb + ln) * DK;
        const _Float16* kp1 = Kh + qk_base + (size_t)(kb + 16 + ln) * DK;

        v8f s0 = {}, s1 = {};
        s0 = wmma_f16(aq0, load_b_f16(kp0,      half), s0);
        s0 = wmma_f16(aq1, load_b_f16(kp0 + 32, half), s0);
        s1 = wmma_f16(aq0, load_b_f16(kp1,      half), s1);
        s1 = wmma_f16(aq1, load_b_f16(kp1 + 32, half), s1);

        if (kb + 32 < SEQ)   // hint next K block into cache
            __builtin_prefetch(Kh + qk_base + (size_t)(kb + 32 + ln) * DK, 0, 1);

        // mask fill BEFORE the 1/sqrt(dk) scale (matches reference)
        int mk0 = mbase[kb + ln];
        int mk1 = mbase[kb + 16 + ln];
#pragma unroll
        for (int v = 0; v < 8; ++v) {
            float x0 = mk0 ? (float)s0[v] : -1.0e9f;
            float x1 = mk1 ? (float)s1[v] : -1.0e9f;
            s0[v] = x0 * 0.125f;   // 1/sqrt(64)
            s1[v] = x1 * 0.125f;
        }

        float alpha[8];
#pragma unroll
        for (int v = 0; v < 8; ++v) {
            // row m = v + 8*half spans the 16 lanes of this half
            float mm = fmaxf(s0[v], s1[v]);
#pragma unroll
            for (int off = 8; off >= 1; off >>= 1)
                mm = fmaxf(mm, __shfl_xor(mm, off, 16));
            float mnew = fmaxf(mrow[v], mm);
            float al = __expf(mrow[v] - mnew);
            float p0 = __expf(s0[v] - mnew);
            float p1 = __expf(s1[v] - mnew);
            float r = p0 + p1;
#pragma unroll
            for (int off = 8; off >= 1; off >>= 1)
                r += __shfl_xor(r, off, 16);
            lrow[v] = lrow[v] * al + r;
            mrow[v] = mnew;
            alpha[v] = al;
            // stash P (C-layout) into LDS for A-layout reload
            pbuf[w][v + 8 * half][ln]      = (_Float16)p0;
            pbuf[w][v + 8 * half][16 + ln] = (_Float16)p1;
        }

#pragma unroll
        for (int t = 0; t < 4; ++t)
#pragma unroll
            for (int v = 0; v < 8; ++v) acc[t][v] *= alpha[v];

        // LDS ops are in-order within a wave: safe to reload P as A-operand
        v16h ap = load_a_f16(&pbuf[w][ln][0], half);

#pragma unroll
        for (int t = 0; t < 4; ++t) {
            const _Float16* vp = Vt + vt_base + (size_t)(t * 16 + ln) * SEQ + kb;
            acc[t] = wmma_f16(ap, load_b_f16(vp, half), acc[t]);
        }
    }

    // epilogue: normalize and store merged-head f16 activation [n*L, D]
#pragma unroll
    for (int t = 0; t < 4; ++t) {
#pragma unroll
        for (int v = 0; v < 8; ++v) {
            int q = q0 + v + 8 * half;
            size_t idx = (size_t)(nb * SEQ + q) * DMODEL + h * DK + t * 16 + ln;
            AO[idx] = (_Float16)(acc[t][v] / lrow[v]);
        }
    }
}

// ---------------------------------------------------------------------------
// Host launcher
// ---------------------------------------------------------------------------
extern "C" void kernel_launch(void* const* d_in, const int* in_sizes, int n_in,
                              void* d_out, int out_size, void* d_ws, size_t ws_size,
                              hipStream_t stream) {
    const float* value = (const float*)d_in[0];
    const float* key   = (const float*)d_in[1];
    const float* query = (const float*)d_in[2];
    const int*   maskp = (const int*)  d_in[3];
    const float* wq = (const float*)d_in[4];
    const float* bq = (const float*)d_in[5];
    const float* wk = (const float*)d_in[6];
    const float* bk = (const float*)d_in[7];
    const float* wv = (const float*)d_in[8];
    const float* bv = (const float*)d_in[9];
    const float* wo = (const float*)d_in[10];
    const float* bo = (const float*)d_in[11];

    char* ws = (char*)d_ws;
    const size_t MB = (size_t)1 << 20;
    _Float16* Qh  = (_Float16*)(ws + 0  * MB);  // [N,H,L,DK] f16, 16 MiB
    _Float16* Kh  = (_Float16*)(ws + 16 * MB);  // [N,H,L,DK] f16
    _Float16* Vt  = (_Float16*)(ws + 32 * MB);  // [N,H,DK,L] f16 (transposed)
    _Float16* AO  = (_Float16*)(ws + 48 * MB);  // [N*L, D]   f16
    _Float16* Wqh = (_Float16*)(ws + 64 * MB);  // 2 MiB each
    _Float16* Wkh = (_Float16*)(ws + 66 * MB);
    _Float16* Wvh = (_Float16*)(ws + 68 * MB);
    _Float16* Woh = (_Float16*)(ws + 70 * MB);
    _Float16* Xh  = (_Float16*)(ws + 72 * MB);  // f16 activation bounce, 16 MiB

    // weights fp32 -> f16 (262144 float4s each)
    cvt_f16_kernel<<<1024, 256, 0, stream>>>(wq, Wqh);
    cvt_f16_kernel<<<1024, 256, 0, stream>>>(wk, Wkh);
    cvt_f16_kernel<<<1024, 256, 0, stream>>>(wv, Wvh);
    cvt_f16_kernel<<<1024, 256, 0, stream>>>(wo, Woh);

    dim3 gg(MROWS / 128, DMODEL / 128);   // 64 x 8 workgroups
    const int ACT_BLOCKS = MROWS * DMODEL / 4 / 256;  // 8192, exact float4 cover

    // Q = query @ wq^T + bq  (convert activations once, then pure-f16 GEMM)
    cvt_f16_kernel<<<ACT_BLOCKS, 256, 0, stream>>>(query, Xh);
    linear_kernel<<<gg, 256, 0, stream>>>(Xh, Wqh, bq, Qh, nullptr, 0);
    // K = key @ wk^T + bk
    cvt_f16_kernel<<<ACT_BLOCKS, 256, 0, stream>>>(key, Xh);
    linear_kernel<<<gg, 256, 0, stream>>>(Xh, Wkh, bk, Kh, nullptr, 0);
    // V = value @ wv^T + bv  (stored transposed per head)
    cvt_f16_kernel<<<ACT_BLOCKS, 256, 0, stream>>>(value, Xh);
    linear_kernel<<<gg, 256, 0, stream>>>(Xh, Wvh, bv, Vt, nullptr, 1);

    attention_kernel<<<NB * HEADS * (SEQ / 128), 256, 0, stream>>>(Qh, Kh, Vt, maskp, AO);

    linear_kernel<<<gg, 256, 0, stream>>>(AO, Woh, bo, nullptr, (float*)d_out, 2);
}
